// Model_32830730011302
// MI455X (gfx1250) — matrix operand order repaired and checked by reading, not compiled, not run on hardware
//
#include <hip/hip_runtime.h>
#include <hip/hip_bf16.h>
#include <math.h>

typedef __bf16 bf16;
typedef bf16  v16bf __attribute__((ext_vector_type(16)));
typedef float v8f   __attribute__((ext_vector_type(8)));

#define B_    16
#define SEQ_  512
#define NF_   21
#define DM_   512
#define NH_   8
#define E_    64
#define DFF_  2048
#define L_    512
#define PRED_ 256
#define LBL_  256

// ---------------------------------------------------------------------------
// Batched WMMA GEMM: C[b] = act(A[b][M,K] * W[b][N,K]^T + bias)
// fp32 in/out, bf16 via LDS. Double-buffered LDS + register prefetch.
// Block: 256 threads = 8 wave32 waves, tile 128(M) x (NFR*32)(N).
// Wave grid 4(M) x 2(N); each wave: 2 x NFR accumulators of 16x16.
// Requires M%128==0, N%(NFR*32)==0, K%32==0.
// act: 0=none, 1=relu, 2=gelu(tanh)
// ---------------------------------------------------------------------------
#define TM 128
#define KT 32
#define KP 40   // padded LDS row (halves)

template<int NFR>
__global__ __launch_bounds__(256) void gemm_wmma_kernel(
    const float* __restrict__ A, long long sA, int lda,
    const float* __restrict__ W, long long sW, int ldw,
    float* __restrict__ C, long long sC, int ldc,
    const float* __restrict__ bias,
    int M, int N, int K, int act)
{
    constexpr int TNt = NFR * 32;     // 64 or 128
    constexpr int WIT = TNt / 32;     // W staging float4 iters (2 or 4)

    __shared__ bf16 As[2][TM][KP];
    __shared__ bf16 Ws[2][TNt][KP];

    const int bz = blockIdx.z;
    A += (long long)bz * sA;
    W += (long long)bz * sW;
    C += (long long)bz * sC;

    const int m0   = blockIdx.y * TM;
    const int n0   = blockIdx.x * TNt;
    const int tid  = threadIdx.x;
    const int wave = tid >> 5;     // 0..7
    const int lane = tid & 31;
    const int wm   = wave >> 1;    // 0..3 -> row offset wm*32
    const int wn   = wave & 1;     // 0..1 -> col offset wn*(NFR*16)
    const int h    = lane >> 4;    // half-wave 0/1
    const int lr   = lane & 15;

    v8f acc[2][NFR];
    const v8f vzero = {0.f,0.f,0.f,0.f,0.f,0.f,0.f,0.f};
    #pragma unroll
    for (int i = 0; i < 2; ++i)
        #pragma unroll
        for (int j = 0; j < NFR; ++j) acc[i][j] = vzero;

    const int r  = tid >> 3;         // 0..31 staging row
    const int c4 = (tid & 7) * 4;    // 0..28 staging col (float4)
    const float* Abase = A + (long long)(m0 + r) * lda + c4;
    const float* Wbase = W + (long long)(n0 + r) * ldw + c4;

    const int nk = K / KT;
    float4 pa[4], pw[WIT];

    // ---- prologue: fetch + stage tile 0 into buffer 0 ----
    #pragma unroll
    for (int it = 0; it < 4; ++it)
        pa[it] = *(const float4*)(Abase + (long long)it * 32 * lda);
    #pragma unroll
    for (int it = 0; it < WIT; ++it)
        pw[it] = *(const float4*)(Wbase + (long long)it * 32 * ldw);
    #pragma unroll
    for (int it = 0; it < 4; ++it) {
        const int row = r + it * 32;
        As[0][row][c4 + 0] = (bf16)pa[it].x; As[0][row][c4 + 1] = (bf16)pa[it].y;
        As[0][row][c4 + 2] = (bf16)pa[it].z; As[0][row][c4 + 3] = (bf16)pa[it].w;
    }
    #pragma unroll
    for (int it = 0; it < WIT; ++it) {
        const int row = r + it * 32;
        Ws[0][row][c4 + 0] = (bf16)pw[it].x; Ws[0][row][c4 + 1] = (bf16)pw[it].y;
        Ws[0][row][c4 + 2] = (bf16)pw[it].z; Ws[0][row][c4 + 3] = (bf16)pw[it].w;
    }
    __syncthreads();

    for (int kt = 0; kt < nk; ++kt) {
        const int buf = kt & 1;

        // Prefetch next K-tile from global while this tile computes
        if (kt + 1 < nk) {
            const int kn = (kt + 1) * KT;
            #pragma unroll
            for (int it = 0; it < 4; ++it)
                pa[it] = *(const float4*)(Abase + (long long)it * 32 * lda + kn);
            #pragma unroll
            for (int it = 0; it < WIT; ++it)
                pw[it] = *(const float4*)(Wbase + (long long)it * 32 * ldw + kn);
        }

        // Fragment loads (ISA 16-bit wave32 layouts)
        v16bf afrag[2], bfrag[NFR];
        #pragma unroll
        for (int i = 0; i < 2; ++i) {
            const int row = wm * 32 + i * 16 + lr;     // A row (M)
            #pragma unroll
            for (int p = 0; p < 8; ++p) {
                const int kk = ((p < 4) ? 2 * p : 2 * p + 8) + 8 * h;
                afrag[i][2 * p]     = As[buf][row][kk];
                afrag[i][2 * p + 1] = As[buf][row][kk + 1];
            }
        }
        #pragma unroll
        for (int j = 0; j < NFR; ++j) {
            const int col = wn * (NFR * 16) + j * 16 + lr;   // B col (N) == W row
            #pragma unroll
            for (int e = 0; e < 16; ++e)
                bfrag[j][e] = Ws[buf][col][((e < 8) ? e : e + 8) + 8 * h];
        }

        #pragma unroll
        for (int i = 0; i < 2; ++i)
            #pragma unroll
            for (int j = 0; j < NFR; ++j)
                acc[i][j] = __builtin_amdgcn_wmma_f32_16x16x32_bf16(
                    false, afrag[i], false, bfrag[j],
                    (short)0, acc[i][j], false, false);

        // Stage the prefetched tile into the other buffer.
        // (WMMA's dscnt wait guarantees our reads of `buf` completed.)
        if (kt + 1 < nk) {
            const int nb = buf ^ 1;
            #pragma unroll
            for (int it = 0; it < 4; ++it) {
                const int row = r + it * 32;
                As[nb][row][c4 + 0] = (bf16)pa[it].x; As[nb][row][c4 + 1] = (bf16)pa[it].y;
                As[nb][row][c4 + 2] = (bf16)pa[it].z; As[nb][row][c4 + 3] = (bf16)pa[it].w;
            }
            #pragma unroll
            for (int it = 0; it < WIT; ++it) {
                const int row = r + it * 32;
                Ws[nb][row][c4 + 0] = (bf16)pw[it].x; Ws[nb][row][c4 + 1] = (bf16)pw[it].y;
                Ws[nb][row][c4 + 2] = (bf16)pw[it].z; Ws[nb][row][c4 + 3] = (bf16)pw[it].w;
            }
            __syncthreads();
        }
    }

    // Epilogue: C VGPR g holds (m = g + 8h, n = lr) of each 16x16 tile
    #pragma unroll
    for (int i = 0; i < 2; ++i) {
        #pragma unroll
        for (int j = 0; j < NFR; ++j) {
            const int n  = n0 + wn * (NFR * 16) + j * 16 + lr;
            const float bv = bias ? bias[n] : 0.f;
            #pragma unroll
            for (int g = 0; g < 8; ++g) {
                const int m = m0 + wm * 32 + i * 16 + g + 8 * h;
                float v = acc[i][j][g] + bv;
                if (act == 1) {
                    v = fmaxf(v, 0.f);
                } else if (act == 2) {
                    const float u = 0.7978845608028654f * (v + 0.044715f * v * v * v);
                    v = 0.5f * v * (1.f + tanhf(u));
                }
                C[(long long)m * ldc + n] = v;
            }
        }
    }
}

// ---------------------------------------------------------------------------
// Per-(b,f) mean/std over time axis
// ---------------------------------------------------------------------------
__global__ void stats_kernel(const float* __restrict__ X,
                             float* __restrict__ meanv, float* __restrict__ stdv)
{
    const int idx = blockIdx.x * blockDim.x + threadIdx.x;
    if (idx >= B_ * NF_) return;
    const int b = idx / NF_, f = idx % NF_;
    const float* p = X + (size_t)b * SEQ_ * NF_ + f;
    float s = 0.f;
    for (int l = 0; l < SEQ_; ++l) s += p[(size_t)l * NF_];
    const float m = s / (float)SEQ_;
    float v = 0.f;
    for (int l = 0; l < SEQ_; ++l) { const float d = p[(size_t)l * NF_] - m; v += d * d; }
    meanv[idx] = m;
    stdv[idx]  = sqrtf(v / (float)SEQ_ + 1e-5f);
}

__global__ void normalize_kernel(const float* __restrict__ X,
                                 const float* __restrict__ meanv,
                                 const float* __restrict__ stdv,
                                 float* __restrict__ xe)
{
    const int n = B_ * SEQ_ * NF_;
    for (int i = blockIdx.x * blockDim.x + threadIdx.x; i < n; i += gridDim.x * blockDim.x) {
        const int f = i % NF_;
        const int b = i / (SEQ_ * NF_);
        xe[i] = (X[i] - meanv[b * NF_ + f]) / stdv[b * NF_ + f];
    }
}

__global__ void build_dec_kernel(const float* __restrict__ xe, float* __restrict__ xdec)
{
    const int n = B_ * L_ * NF_;
    for (int i = blockIdx.x * blockDim.x + threadIdx.x; i < n; i += gridDim.x * blockDim.x) {
        const int f = i % NF_;
        const int l = (i / NF_) % L_;
        const int b = i / (L_ * NF_);
        xdec[i] = (l < LBL_) ? xe[((size_t)b * SEQ_ + (SEQ_ - LBL_) + l) * NF_ + f] : 0.f;
    }
}

// ---------------------------------------------------------------------------
// Projector conv (x_raw [B,SEQ,NF] as NCH with C=SEQ, circular over NF, k=3)
// ---------------------------------------------------------------------------
__global__ void proj_conv_kernel(const float* __restrict__ X,
                                 const float* __restrict__ w,  // [SEQ*3]
                                 float* __restrict__ out)      // [B,NF]
{
    const int b = blockIdx.x;
    const int f = threadIdx.x;
    if (f >= NF_) return;
    const int fm = (f + NF_ - 1) % NF_;
    const int fp = (f + 1) % NF_;
    float s = 0.f;
    for (int c = 0; c < SEQ_; ++c) {
        const float* xr = X + ((size_t)b * SEQ_ + c) * NF_;
        const float* wr = w + c * 3;
        s += wr[0] * xr[fm] + wr[1] * xr[f] + wr[2] * xr[fp];
    }
    out[b * NF_ + f] = s;
}

// ---------------------------------------------------------------------------
// Projector MLP: concat(conv, stats)[42] -> 128 relu -> 128 relu -> out_dim
// ---------------------------------------------------------------------------
__global__ __launch_bounds__(128) void proj_mlp_kernel(
    const float* __restrict__ pconv, const float* __restrict__ stats,
    const float* __restrict__ W0, const float* __restrict__ b0,
    const float* __restrict__ W1, const float* __restrict__ b1,
    const float* __restrict__ W2,
    float* __restrict__ out, int out_dim, int do_exp)
{
    const int b = blockIdx.x;
    const int t = threadIdx.x;
    __shared__ float hin[2 * NF_], h0[128], h1[128];
    if (t < NF_)            hin[t] = pconv[b * NF_ + t];
    else if (t < 2 * NF_)   hin[t] = stats[b * NF_ + (t - NF_)];
    __syncthreads();
    float s = b0[t];
    for (int i = 0; i < 2 * NF_; ++i) s += W0[t * (2 * NF_) + i] * hin[i];
    h0[t] = fmaxf(s, 0.f);
    __syncthreads();
    s = b1[t];
    for (int i = 0; i < 128; ++i) s += W1[t * 128 + i] * h0[i];
    h1[t] = fmaxf(s, 0.f);
    __syncthreads();
    for (int o = t; o < out_dim; o += 128) {
        float v = 0.f;
        for (int i = 0; i < 128; ++i) v += W2[o * 128 + i] * h1[i];
        out[b * out_dim + o] = do_exp ? expf(v) : v;
    }
}

// ---------------------------------------------------------------------------
// Token embedding (circular conv k=3 over time, NF->DM) + sinusoidal PE
// ---------------------------------------------------------------------------
__global__ __launch_bounds__(512) void embed_kernel(
    const float* __restrict__ X,   // [B,L,NF]
    const float* __restrict__ Wc,  // [DM,NF,3]
    float* __restrict__ O)         // [B*L,DM]
{
    const int bl = blockIdx.x;
    const int b = bl / L_, l = bl % L_;
    __shared__ float xs[3][NF_];
    const int t = threadIdx.x;
    if (t < 3 * NF_) {
        const int j = t / NF_, f = t % NF_;
        const int lc = (l + j - 1 + L_) % L_;
        xs[j][f] = X[((size_t)b * L_ + lc) * NF_ + f];
    }
    __syncthreads();
    const int d = t;
    float s = 0.f;
    #pragma unroll 3
    for (int j = 0; j < 3; ++j)
        for (int f = 0; f < NF_; ++f)
            s += Wc[(d * NF_ + f) * 3 + j] * xs[j][f];
    const float i2 = (float)((d >> 1) * 2);
    const float freq = __expf(-i2 * 9.210340371976184f / (float)DM_); // ln(1e4)
    const float ang = (float)l * freq;
    s += (d & 1) ? __cosf(ang) : __sinf(ang);
    O[(size_t)bl * DM_ + d] = s;
}

// ---------------------------------------------------------------------------
// Head split: Q,K -> [B,NH,L,E]; V -> [B,NH,E,L] (transposed for PV GEMM)
// ---------------------------------------------------------------------------
__global__ void split_qkv_kernel(const float* __restrict__ Q,
                                 const float* __restrict__ K,
                                 const float* __restrict__ V,
                                 float* __restrict__ qh, float* __restrict__ kh,
                                 float* __restrict__ vt)
{
    const int n = B_ * NH_ * L_ * E_;
    for (int i = blockIdx.x * blockDim.x + threadIdx.x; i < n; i += gridDim.x * blockDim.x) {
        const int e = i & (E_ - 1);
        const int l = (i >> 6) & (L_ - 1);
        const int h = (i >> 15) & (NH_ - 1);
        const int b = i >> 18;
        const size_t src = ((size_t)(b * L_ + l)) * DM_ + h * E_ + e;
        qh[i] = Q[src];
        kh[i] = K[src];
        vt[(((size_t)(b * NH_ + h) * E_ + e) << 9) + l] = V[src];
    }
}

__global__ void merge_heads_kernel(const float* __restrict__ HO, float* __restrict__ Mg)
{
    const int n = B_ * NH_ * L_ * E_;
    for (int i = blockIdx.x * blockDim.x + threadIdx.x; i < n; i += gridDim.x * blockDim.x) {
        const int e = i & (E_ - 1);
        const int l = (i >> 6) & (L_ - 1);
        const int h = (i >> 15) & (NH_ - 1);
        const int b = i >> 18;
        Mg[((size_t)(b * L_ + l)) * DM_ + h * E_ + e] = HO[i];
    }
}

// ---------------------------------------------------------------------------
// Softmax over S=512 with tau scaling, optional delta, optional causal mask.
// One block per row (h*L + l) of the per-b chunk.
// ---------------------------------------------------------------------------
__global__ __launch_bounds__(256) void softmax_kernel(
    float* __restrict__ S, const float* __restrict__ tauv,
    const float* __restrict__ delta, int causal)
{
    const int row = blockIdx.x;
    const int l = row & (L_ - 1);
    float* p = S + (size_t)row * L_;
    const int t = threadIdx.x;
    __shared__ float red[256];
    const float tv = tauv[0];
    float vals[2];
    float mx = -1e30f;
    #pragma unroll
    for (int it = 0; it < 2; ++it) {
        const int s = t + it * 256;
        float v = p[s] * tv + (delta ? delta[s] : 0.f);
        if (causal && s > l) v = -1e30f;
        v *= 0.125f;   // 1/sqrt(E=64)
        vals[it] = v;
        mx = fmaxf(mx, v);
    }
    red[t] = mx; __syncthreads();
    for (int o = 128; o > 0; o >>= 1) { if (t < o) red[t] = fmaxf(red[t], red[t + o]); __syncthreads(); }
    mx = red[0]; __syncthreads();
    float sum = 0.f;
    #pragma unroll
    for (int it = 0; it < 2; ++it) { vals[it] = __expf(vals[it] - mx); sum += vals[it]; }
    red[t] = sum; __syncthreads();
    for (int o = 128; o > 0; o >>= 1) { if (t < o) red[t] += red[t + o]; __syncthreads(); }
    const float inv = 1.f / red[0];
    #pragma unroll
    for (int it = 0; it < 2; ++it) p[t + it * 256] = vals[it] * inv;
}

// ---------------------------------------------------------------------------
// out = LayerNorm(X + Y) * g + b   (Y may be null). D = 512. One block per row.
// ---------------------------------------------------------------------------
__global__ __launch_bounds__(256) void add_ln_kernel(
    const float* __restrict__ X, const float* __restrict__ Y,
    const float* __restrict__ g, const float* __restrict__ bb,
    float* __restrict__ O)
{
    const int row = blockIdx.x;
    const int t = threadIdx.x;
    const size_t base = (size_t)row * DM_;
    __shared__ float red[256];
    float v0 = X[base + t]       + (Y ? Y[base + t]       : 0.f);
    float v1 = X[base + t + 256] + (Y ? Y[base + t + 256] : 0.f);
    red[t] = v0 + v1; __syncthreads();
    for (int o = 128; o > 0; o >>= 1) { if (t < o) red[t] += red[t + o]; __syncthreads(); }
    const float m = red[0] * (1.f / (float)DM_); __syncthreads();
    const float d0 = v0 - m, d1 = v1 - m;
    red[t] = d0 * d0 + d1 * d1; __syncthreads();
    for (int o = 128; o > 0; o >>= 1) { if (t < o) red[t] += red[t + o]; __syncthreads(); }
    const float inv = rsqrtf(red[0] * (1.f / (float)DM_) + 1e-5f);
    O[base + t]       = d0 * inv * g[t]       + bb[t];
    O[base + t + 256] = d1 * inv * g[t + 256] + bb[t + 256];
}

// ---------------------------------------------------------------------------
// Final: out[b,t,f] = (dec[b, LBL+t] . Wout[f] + bout[f]) * std + mean
// ---------------------------------------------------------------------------
__global__ __launch_bounds__(64) void final_kernel(
    const float* __restrict__ dec, const float* __restrict__ Wout,
    const float* __restrict__ bout, const float* __restrict__ stdv,
    const float* __restrict__ meanv, float* __restrict__ out)
{
    const int bt = blockIdx.x;
    const int b = bt / PRED_, t = bt % PRED_;
    const int l = LBL_ + t;
    __shared__ float dr[DM_];
    for (int i = threadIdx.x; i < DM_; i += 64)
        dr[i] = dec[((size_t)b * L_ + l) * DM_ + i];
    __syncthreads();
    if (threadIdx.x < NF_) {
        const int f = threadIdx.x;
        float v = bout[f];
        for (int i = 0; i < DM_; ++i) v += dr[i] * Wout[f * DM_ + i];
        out[((size_t)b * PRED_ + t) * NF_ + f] = v * stdv[b * NF_ + f] + meanv[b * NF_ + f];
    }
}

// ===========================================================================
// Host orchestration
// ===========================================================================
extern "C" void kernel_launch(void* const* d_in, const int* in_sizes, int n_in,
                              void* d_out, int out_size, void* d_ws, size_t ws_size,
                              hipStream_t stream)
{
    auto P = [&](int i) { return (const float*)d_in[i]; };

    // Parameter leaf indices (recursive insertion order of setup_inputs()):
    const int I_XENC = 0, I_ENCEMB = 2, I_DECEMB = 3;
    const int I_TAU = 4, I_DELTA = 10;                  // conv,W0,b0,W1,b1,W2
    const int I_ENCL = 16;                              // 16 leaves per enc layer
    const int I_ENCNORM = 48;
    const int I_DECL = 50;                              // sattn(8) cattn(8) ffn(4) n1..n3(6)
    const int I_DECNORM = 76, I_WOUT = 78, I_BOUT = 79;

    // Workspace layout (floats)
    float* w = (float*)d_ws;
    size_t off = 0;
    auto alloc = [&](size_t n) { float* p = w + off; off += (n + 63) & ~(size_t)63; return p; };
    const size_t ACT = (size_t)B_ * L_ * DM_;           // 4,194,304
    float* xe    = alloc((size_t)B_ * SEQ_ * NF_);
    float* xdec  = alloc((size_t)B_ * L_ * NF_);
    float* meanv = alloc(B_ * NF_);
    float* stdv  = alloc(B_ * NF_);
    float* pconv = alloc(B_ * NF_);
    float* tau   = alloc(B_);
    float* delta = alloc((size_t)B_ * SEQ_);
    float* X0  = alloc(ACT);
    float* X1  = alloc(ACT);
    float* Qb  = alloc(ACT);
    float* Kb  = alloc(ACT);
    float* Vb  = alloc(ACT);
    float* QH  = alloc(ACT);
    float* KH  = alloc(ACT);
    float* VT  = alloc(ACT);
    float* HO  = alloc(ACT);
    float* XN  = alloc(ACT);
    float* ENC = alloc(ACT);
    float* FFH = alloc((size_t)B_ * L_ * DFF_);
    float* SC  = alloc((size_t)NH_ * L_ * L_);
    if (off * sizeof(float) > ws_size) return;

    const int M = B_ * L_;  // 8192

    // nfr=4 -> 128-wide tiles (all N%128==0 GEMMs); nfr=2 -> 64-wide (PV, N=64)
    auto gemm = [&](const float* A, long long sA, int lda,
                    const float* Wm, long long sW, int ldw,
                    float* C, long long sC, int ldc,
                    const float* bias, int Mm, int Nn, int Kk, int act, int batch,
                    int nfr) {
        if (nfr == 4) {
            dim3 g(Nn / 128, Mm / TM, batch);
            gemm_wmma_kernel<4><<<g, 256, 0, stream>>>(A, sA, lda, Wm, sW, ldw,
                                                       C, sC, ldc, bias, Mm, Nn, Kk, act);
        } else {
            dim3 g(Nn / 64, Mm / TM, batch);
            gemm_wmma_kernel<2><<<g, 256, 0, stream>>>(A, sA, lda, Wm, sW, ldw,
                                                       C, sC, ldc, bias, Mm, Nn, Kk, act);
        }
    };

    // attention(out=X1): q from Xq, k/v from Xkv; pbase -> Wq,bq,Wk,bk,Wv,bv,Wo,bo
    auto attention = [&](const float* Xq, const float* Xkv, int pbase,
                         const float* deltaAll, int causal) {
        gemm(Xq,  0, DM_, P(pbase + 0), 0, DM_, Qb, 0, DM_, P(pbase + 1), M, DM_, DM_, 0, 1, 4);
        gemm(Xkv, 0, DM_, P(pbase + 2), 0, DM_, Kb, 0, DM_, P(pbase + 3), M, DM_, DM_, 0, 1, 4);
        gemm(Xkv, 0, DM_, P(pbase + 4), 0, DM_, Vb, 0, DM_, P(pbase + 5), M, DM_, DM_, 0, 1, 4);
        split_qkv_kernel<<<4096, 256, 0, stream>>>(Qb, Kb, Vb, QH, KH, VT);
        const long long HS = (long long)L_ * E_;        // per-head stride (q/k/vt)
        const long long SS = (long long)L_ * L_;        // per-head score stride
        for (int b = 0; b < B_; ++b) {
            const size_t bo = (size_t)b * NH_ * L_ * E_;
            gemm(QH + bo, HS, E_, KH + bo, HS, E_, SC, SS, L_,
                 nullptr, L_, L_, E_, 0, NH_, 4);
            softmax_kernel<<<NH_ * L_, 256, 0, stream>>>(
                SC, tau + b, deltaAll ? (deltaAll + (size_t)b * SEQ_) : nullptr, causal);
            gemm(SC, SS, L_, VT + bo, HS, L_, HO + bo, HS, E_,
                 nullptr, L_, E_, L_, 0, NH_, 2);
        }
        merge_heads_kernel<<<4096, 256, 0, stream>>>(HO, Qb);
        gemm(Qb, 0, DM_, P(pbase + 6), 0, DM_, X1, 0, DM_, P(pbase + 7), M, DM_, DM_, 0, 1, 4);
    };

    auto ffn = [&](const float* Xin, int fbase, float* Out) {
        gemm(Xin, 0, DM_, P(fbase + 0), 0, DM_, FFH, 0, DFF_, P(fbase + 1), M, DFF_, DM_, 2, 1, 4);
        gemm(FFH, 0, DFF_, P(fbase + 2), 0, DFF_, Out, 0, DM_, P(fbase + 3), M, DM_, DFF_, 0, 1, 4);
    };

    // ---- Phase 0: stats, normalization, projectors --------------------------
    stats_kernel<<<3, 128, 0, stream>>>(P(I_XENC), meanv, stdv);
    normalize_kernel<<<1024, 256, 0, stream>>>(P(I_XENC), meanv, stdv, xe);
    build_dec_kernel<<<1024, 256, 0, stream>>>(xe, xdec);

    proj_conv_kernel<<<B_, 32, 0, stream>>>(P(I_XENC), P(I_TAU + 0), pconv);
    proj_mlp_kernel<<<B_, 128, 0, stream>>>(pconv, stdv,
        P(I_TAU + 1), P(I_TAU + 2), P(I_TAU + 3), P(I_TAU + 4), P(I_TAU + 5),
        tau, 1, 1);
    proj_conv_kernel<<<B_, 32, 0, stream>>>(P(I_XENC), P(I_DELTA + 0), pconv);
    proj_mlp_kernel<<<B_, 128, 0, stream>>>(pconv, meanv,
        P(I_DELTA + 1), P(I_DELTA + 2), P(I_DELTA + 3), P(I_DELTA + 4), P(I_DELTA + 5),
        delta, SEQ_, 0);

    // ---- Phase 1: encoder ---------------------------------------------------
    embed_kernel<<<B_ * L_, DM_, 0, stream>>>(xe, P(I_ENCEMB), X0);
    for (int li = 0; li < 2; ++li) {
        const int lb = I_ENCL + li * 16;   // attn(8), ffn(4), n1(2), n2(2)
        attention(X0, X0, lb + 0, delta, 0);
        add_ln_kernel<<<M, 256, 0, stream>>>(X0, X1, P(lb + 12), P(lb + 13), XN);
        ffn(XN, lb + 8, X1);
        add_ln_kernel<<<M, 256, 0, stream>>>(XN, X1, P(lb + 14), P(lb + 15), X0);
    }
    add_ln_kernel<<<M, 256, 0, stream>>>(X0, nullptr, P(I_ENCNORM), P(I_ENCNORM + 1), ENC);

    // ---- Phase 2: decoder ---------------------------------------------------
    embed_kernel<<<B_ * L_, DM_, 0, stream>>>(xdec, P(I_DECEMB), X0);
    {
        const int lb = I_DECL;   // sattn(8) cattn(8) ffn(4) n1 n2 n3
        attention(X0, X0, lb + 0, nullptr, 1);                       // causal self-attn
        add_ln_kernel<<<M, 256, 0, stream>>>(X0, X1, P(lb + 20), P(lb + 21), XN);
        attention(XN, ENC, lb + 8, delta, 0);                        // cross-attn
        add_ln_kernel<<<M, 256, 0, stream>>>(XN, X1, P(lb + 22), P(lb + 23), X0);
        ffn(X0, lb + 16, X1);
        add_ln_kernel<<<M, 256, 0, stream>>>(X0, X1, P(lb + 24), P(lb + 25), XN);
    }
    add_ln_kernel<<<M, 256, 0, stream>>>(XN, nullptr, P(I_DECNORM), P(I_DECNORM + 1), X0);

    final_kernel<<<B_ * PRED_, 64, 0, stream>>>(X0, P(I_WOUT), P(I_BOUT),
                                                stdv, meanv, (float*)d_out);
}
